// BiDAFAttention_71365176590348
// MI455X (gfx1250) — compile-verified, matrix operand order checked by reading, not run on hardware
//
#include <hip/hip_runtime.h>

// BiDAF attention for MI455X (gfx1250, wave32).
// Fused S -> masked softmax -> c2q in one kernel using v_wmma_f32_16x16x32_bf16.
// Memory-bound workload (~90MB traffic, ~2.2 GFLOP): WMMA removes the compute wall.

#define B_   16
#define CL   2048
#define QL   128
#define HD   128
#define NEGF -1000000000.0f
#define QP   132   // query LDS pitch in halves (even, padded vs 128 to spread banks)
#define AP   132   // prob  LDS pitch in halves
#define NW   4     // waves per block in kernel 1 (64 context rows per block)

typedef __attribute__((ext_vector_type(16))) __bf16 v16bf;
typedef __attribute__((ext_vector_type(8)))  float  v8f;

union BF16Frag { v16bf v; unsigned int u[8]; };

static __device__ __forceinline__ unsigned short f2bf(float f) {
  unsigned int u = __float_as_uint(f);
  u += 0x7FFFu + ((u >> 16) & 1u);   // round-to-nearest-even
  return (unsigned short)(u >> 16);
}

// 16-bit A/B fragment K pattern (ISA 7.12.2, 16-bit A 16x32):
// lane group hi = lane>>4 adds +8 to K; vgpr j holds K pair (j<4 ? 2j : 2j+8).
static __device__ __forceinline__ int kpat(int j, int hi, int kOff) {
  return kOff + 8 * hi + (j < 4 ? 2 * j : 8 + 2 * j);
}

__global__ __launch_bounds__(32 * NW) void bidaf_scores_kernel(
    const float* __restrict__ ctx, const float* __restrict__ qry,
    const float* __restrict__ W, const int* __restrict__ cmask,
    const int* __restrict__ qmask, float* __restrict__ G,
    float* __restrict__ maxq_ws)
{
  __shared__ unsigned short qlds[QL * QP];        // query, bf16, row-major
  __shared__ unsigned short abuf[NW * 16 * AP];   // per-wave softmax probs, bf16
  __shared__ float wlds[3 * HD];
  __shared__ float sq[QL];
  __shared__ int   qm[QL];
  __shared__ float scbuf[NW * 16];

  const int tid  = threadIdx.x;
  const int wid  = tid >> 5;
  const int lane = tid & 31;
  const int lo   = lane & 15;
  const int hi   = lane >> 4;

  const int b      = blockIdx.x >> 5;         // 32 context blocks per batch
  const int cblk   = blockIdx.x & 31;
  const int base_c = cblk * (16 * NW) + wid * 16;

  // ---- stage W and query(bf16) into LDS ----
  for (int i = tid; i < 3 * HD; i += blockDim.x) wlds[i] = W[i];
  for (int i = tid; i < QL * HD; i += blockDim.x) {
    int q = i >> 7, h = i & (HD - 1);
    qlds[q * QP + h] = f2bf(qry[(size_t)b * QL * HD + i]);
  }
  __syncthreads();

  // ---- s_q = query . w_q, query mask ----
  for (int q = tid; q < QL; q += blockDim.x) {
    const float* row = qry + ((size_t)b * QL + q) * HD;
    float acc = 0.f;
    for (int h = 0; h < HD; ++h) acc += row[h] * wlds[HD + h];
    sq[q] = acc;
    qm[q] = qmask[b * QL + q];
  }

  // ---- s_c = context . w_c for this wave's 16 rows ----
  const float* ctxb = ctx + ((size_t)b * CL + base_c) * HD;
  for (int r = 0; r < 16; ++r) {
    const float* row = ctxb + r * HD;
    float part = 0.f;
    for (int h = lane; h < HD; h += 32) part += row[h] * wlds[h];
#pragma unroll
    for (int off = 16; off > 0; off >>= 1) part += __shfl_xor(part, off, 32);
    if (lane == 0) scbuf[wid * 16 + r] = part;
  }
  __syncthreads();

  // ---- Phase 2: S = (w_cq * context) @ query^T  (16 x 128), 32 WMMAs ----
  v8f acc[8];
#pragma unroll
  for (int t = 0; t < 8; ++t)
#pragma unroll
    for (int v = 0; v < 8; ++v) acc[t][v] = 0.0f;

  const float* wcq = wlds + 2 * HD;
#pragma unroll
  for (int kk = 0; kk < 4; ++kk) {
    BF16Frag af;   // A: row = lo of this wave's tile, K = h
    {
      const float* p = ctxb + lo * HD;
#pragma unroll
      for (int j = 0; j < 8; ++j) {
        int k = kpat(j, hi, 32 * kk);
        float a0 = p[k]     * wcq[k];
        float a1 = p[k + 1] * wcq[k + 1];
        af.u[j] = (unsigned int)f2bf(a0) | ((unsigned int)f2bf(a1) << 16);
      }
    }
#pragma unroll
    for (int t = 0; t < 8; ++t) {
      BF16Frag bq;  // B col n=lo -> query row 16t+lo, K = h (contiguous pairs)
      const unsigned short* qrow = qlds + (16 * t + lo) * QP;
#pragma unroll
      for (int j = 0; j < 8; ++j)
        bq.u[j] = *(const unsigned int*)(qrow + kpat(j, hi, 32 * kk));
      acc[t] = __builtin_amdgcn_wmma_f32_16x16x32_bf16(
          false, af.v, false, bq.v, (short)0, acc[t], false, false);
    }
  }

  // ---- Phase 3: bias add, query mask, row max, softmax (registers + shfl) ----
  // C/D layout: element (vgpr v, lane) = row M = v + 8*hi, col q = 16t + lo.
  float scv[8];
#pragma unroll
  for (int v = 0; v < 8; ++v) scv[v] = scbuf[wid * 16 + 8 * hi + v];

  float rmax[8];
#pragma unroll
  for (int v = 0; v < 8; ++v) rmax[v] = -3.0e38f;

#pragma unroll
  for (int t = 0; t < 8; ++t) {
    float sqv = sq[16 * t + lo];
    int   mq  = qm[16 * t + lo];
#pragma unroll
    for (int v = 0; v < 8; ++v) {
      float s = mq ? (acc[t][v] + scv[v] + sqv) : NEGF;
      acc[t][v] = s;
      rmax[v] = fmaxf(rmax[v], s);
    }
  }
#pragma unroll
  for (int off = 1; off < 16; off <<= 1)
#pragma unroll
    for (int v = 0; v < 8; ++v)
      rmax[v] = fmaxf(rmax[v], __shfl_xor(rmax[v], off, 32));

  // context-masked row max for the q2c softmax (kernel 2)
  if (lo == 0) {
#pragma unroll
    for (int v = 0; v < 8; ++v) {
      int c = base_c + 8 * hi + v;
      maxq_ws[b * CL + c] = cmask[b * CL + c] ? rmax[v] : NEGF;
    }
  }

  float rsum[8];
#pragma unroll
  for (int v = 0; v < 8; ++v) rsum[v] = 0.f;
#pragma unroll
  for (int t = 0; t < 8; ++t)
#pragma unroll
    for (int v = 0; v < 8; ++v) {
      float e = __expf(acc[t][v] - rmax[v]);
      acc[t][v] = e;
      rsum[v] += e;
    }
#pragma unroll
  for (int off = 1; off < 16; off <<= 1)
#pragma unroll
    for (int v = 0; v < 8; ++v)
      rsum[v] += __shfl_xor(rsum[v], off, 32);

  // stage probabilities (bf16) for the c2q A fragments
  unsigned short* aw = abuf + wid * 16 * AP;
#pragma unroll
  for (int v = 0; v < 8; ++v) {
    float rinv = 1.0f / rsum[v];
#pragma unroll
    for (int t = 0; t < 8; ++t)
      aw[(8 * hi + v) * AP + 16 * t + lo] = f2bf(acc[t][v] * rinv);
  }
  __syncthreads();

  // ---- Phase 4: c2q = a @ query  (16 x 128), K = q, 32 WMMAs ----
  v8f acc2[8];
#pragma unroll
  for (int t = 0; t < 8; ++t)
#pragma unroll
    for (int v = 0; v < 8; ++v) acc2[t][v] = 0.0f;

#pragma unroll
  for (int kk = 0; kk < 4; ++kk) {
    BF16Frag af;   // A: probs row lo, K = q (contiguous pairs in abuf)
    const unsigned short* arow = aw + lo * AP;
#pragma unroll
    for (int j = 0; j < 8; ++j)
      af.u[j] = *(const unsigned int*)(arow + kpat(j, hi, 32 * kk));
#pragma unroll
    for (int t = 0; t < 8; ++t) {
      BF16Frag bq;  // B col n=lo -> h = 16t+lo, K = q (column gather from qlds)
#pragma unroll
      for (int j = 0; j < 8; ++j) {
        int k = kpat(j, hi, 32 * kk);
        unsigned int e0 = qlds[(k    ) * QP + 16 * t + lo];
        unsigned int e1 = qlds[(k + 1) * QP + 16 * t + lo];
        bq.u[j] = e0 | (e1 << 16);
      }
      acc2[t] = __builtin_amdgcn_wmma_f32_16x16x32_bf16(
          false, af.v, false, bq.v, (short)0, acc2[t], false, false);
    }
  }

  // ---- Phase 5: write G[..., 0:3H] = [context, c2q, context*c2q] ----
#pragma unroll
  for (int v = 0; v < 8; ++v) {
    int c = base_c + 8 * hi + v;
    const float* crow = ctx + ((size_t)b * CL + c) * HD;
    float*       grow = G + ((size_t)b * CL + c) * (4 * HD);
#pragma unroll
    for (int t = 0; t < 8; ++t) {
      int h = 16 * t + lo;
      float cv  = crow[h];
      float c2q = acc2[t][v];
      grow[h]          = cv;
      grow[HD + h]     = c2q;
      grow[2 * HD + h] = cv * c2q;
    }
  }
}

__global__ __launch_bounds__(256) void bidaf_q2c_kernel(
    const float* __restrict__ ctx, const float* __restrict__ maxq,
    float* __restrict__ q2c)
{
  __shared__ float red[256];
  __shared__ float ew[CL];
  const int b = blockIdx.x, tid = threadIdx.x;

  float m = -3.0e38f;
  for (int c = tid; c < CL; c += 256) m = fmaxf(m, maxq[b * CL + c]);
  red[tid] = m; __syncthreads();
  for (int s = 128; s > 0; s >>= 1) {
    if (tid < s) red[tid] = fmaxf(red[tid], red[tid + s]);
    __syncthreads();
  }
  float M = red[0];
  __syncthreads();

  float s = 0.f;
  for (int c = tid; c < CL; c += 256) {
    float e = __expf(maxq[b * CL + c] - M);
    ew[c] = e;
    s += e;
  }
  red[tid] = s; __syncthreads();
  for (int st = 128; st > 0; st >>= 1) {
    if (tid < st) red[tid] += red[tid + st];
    __syncthreads();
  }
  float tot = red[0];
  __syncthreads();

  if (tid < HD) {
    float acc = 0.f;
    const float* base = ctx + (size_t)b * CL * HD + tid;
    for (int c = 0; c < CL; ++c) acc += ew[c] * base[(size_t)c * HD];
    q2c[b * HD + tid] = acc / tot;
  }
}

__global__ __launch_bounds__(256) void bidaf_g4_kernel(
    const float* __restrict__ ctx, const float* __restrict__ q2c,
    float* __restrict__ G)
{
  size_t i = (size_t)blockIdx.x * blockDim.x + threadIdx.x;  // over B*CL*HD
  int b   = (int)(i >> 18);                                  // CL*HD = 2^18
  int rem = (int)(i & ((1u << 18) - 1u));
  int c = rem >> 7;
  int h = rem & (HD - 1);
  G[(((size_t)b * CL + c) * (4 * HD)) + 3 * HD + h] = ctx[i] * q2c[b * HD + h];
}

extern "C" void kernel_launch(void* const* d_in, const int* in_sizes, int n_in,
                              void* d_out, int out_size, void* d_ws, size_t ws_size,
                              hipStream_t stream) {
  const float* ctx   = (const float*)d_in[0];
  const float* qry   = (const float*)d_in[1];
  const float* W     = (const float*)d_in[2];
  const int*   cmask = (const int*)d_in[3];
  const int*   qmask = (const int*)d_in[4];
  float* G = (float*)d_out;

  float* maxq = (float*)d_ws;          // B*CL floats
  float* q2c  = maxq + B_ * CL;        // B*HD floats

  bidaf_scores_kernel<<<B_ * (CL / (16 * NW)), 32 * NW, 0, stream>>>(
      ctx, qry, W, cmask, qmask, G, maxq);
  bidaf_q2c_kernel<<<B_, 256, 0, stream>>>(ctx, maxq, q2c);
  bidaf_g4_kernel<<<(B_ * CL * HD) / 256, 256, 0, stream>>>(ctx, q2c, G);
}
